// Method_40913858461977
// MI455X (gfx1250) — compile-verified, matrix operand order checked by reading, not run on hardware
//
#include <hip/hip_runtime.h>
#include <hip/hip_bf16.h>
#include <stdint.h>

// ---------------------------------------------------------------------------
// Problem constants (T=512, B=64, HD=ID=1024)
// ---------------------------------------------------------------------------
#define T_STEPS 512
#define B_DIM   64
#define HD_DIM  1024
#define K_DIM   1024
#define NWG     32              // persistent workgroups for the scan
#define HS      (HD_DIM / NWG)  // 32 columns per workgroup

// Padded LDS row stride (elements). 1032 bf16 = 2064 bytes; 2064 % 256 == 16,
// so row j starts 4 banks after row j-1 -> the 16 lanes of a b128 fragment
// load land on 16 disjoint 4-bank groups (conflict-free).
#define LDSW_STRIDE 1032
#define SCAN_LDS_BYTES (2u * 32u * LDSW_STRIDE * 2u)   // 132096 B (F + R regions)

typedef __attribute__((ext_vector_type(16))) __bf16 v16bf;
typedef __attribute__((ext_vector_type(8)))  float  v8f;

__device__ __forceinline__ float sigmoidf_(float v) {
    return 1.0f / (1.0f + __expf(-v));
}

// WMMA D = A(16x32 bf16) * B(32x16 bf16) + C(16x16 f32)
__device__ __forceinline__ v8f wmma_bf16(v16bf a, v16bf b, v8f c) {
    return __builtin_amdgcn_wmma_f32_16x16x32_bf16(
        /*neg_a=*/false, a, /*neg_b=*/false, b,
        /*c_mod=*/(short)0, c, /*reuse_a=*/false, /*reuse_b=*/false);
}

// Load one 16-bit A/B fragment (16 bf16 per lane) from a row-major row.
// Lane k-pattern per ISA 7.12.2: lanes 0-15 hold K {0..7,16..23}, lanes 16-31
// hold K {8..15,24..31}; caller passes off1 = k0+kb1, off2 = k0+kb2.
__device__ __forceinline__ v16bf ld_frag_bf16(const __bf16* row, int off1, int off2) {
    union { uint4 u[2]; v16bf v; } t;
    t.u[0] = *reinterpret_cast<const uint4*>(row + off1);
    t.u[1] = *reinterpret_cast<const uint4*>(row + off2);
    return t.v;
}

// Same fragment but sourced from fp32 (convert on the fly; used for x).
__device__ __forceinline__ v16bf ld_frag_f32(const float* row, int off1, int off2) {
    float4 a0 = *reinterpret_cast<const float4*>(row + off1);
    float4 a1 = *reinterpret_cast<const float4*>(row + off1 + 4);
    float4 b0 = *reinterpret_cast<const float4*>(row + off2);
    float4 b1 = *reinterpret_cast<const float4*>(row + off2 + 4);
    v16bf v;
    v[0] = (__bf16)a0.x; v[1] = (__bf16)a0.y; v[2]  = (__bf16)a0.z; v[3]  = (__bf16)a0.w;
    v[4] = (__bf16)a1.x; v[5] = (__bf16)a1.y; v[6]  = (__bf16)a1.z; v[7]  = (__bf16)a1.w;
    v[8] = (__bf16)b0.x; v[9] = (__bf16)b0.y; v[10] = (__bf16)b0.z; v[11] = (__bf16)b0.w;
    v[12]= (__bf16)b1.x; v[13]= (__bf16)b1.y; v[14] = (__bf16)b1.z; v[15] = (__bf16)b1.w;
    return v;
}

// ---------------------------------------------------------------------------
// Tensor Data Mover: iterate-mode row copy Global -> LDS with padded LDS
// stride (descriptor layout per CDNA5 ISA).
// This toolchain exposes the 6-arg builtin:
//   (uint32x4 g0, int32x8 g1, int32x4 g2, int32x4 g3, int32x8 g4, i32 cpol)
// ---------------------------------------------------------------------------
#if defined(__HIP_DEVICE_COMPILE__) && __has_builtin(__builtin_amdgcn_tensor_load_to_lds)
#define HAVE_TDM 1
typedef __attribute__((ext_vector_type(4))) unsigned int v4u;
typedef __attribute__((ext_vector_type(8))) int          v8i;
typedef __attribute__((ext_vector_type(4))) int          v4i;

__device__ __forceinline__ void tdm_load_rows(unsigned lds_off_bytes,
                                              const __bf16* gsrc,
                                              unsigned nrows,
                                              unsigned rowlen_elts,
                                              unsigned grow_stride_elts,
                                              unsigned lds_stride_elts) {
    unsigned long long ga = (unsigned long long)(uintptr_t)gsrc;
    v4u g0;
    g0[0] = 1u;                                        // count=1 (valid user D#)
    g0[1] = lds_off_bytes;                             // lds_addr
    g0[2] = (unsigned)(ga & 0xFFFFFFFFull);            // global_addr[31:0]
    g0[3] = (unsigned)((ga >> 32) & 0x01FFFFFFull)     // global_addr[56:32]
          | (2u << 30);                                // type = 2 ("image")
    v8i g1;
    g1[0] = (int)((1u << 16) |                         // data_size = 1 -> 2 bytes
                  (1u << 19));                         // iterate_enable
    g1[1] = (int)((rowlen_elts & 0xFFFFu) << 16);      // tensor_dim0[15:0]
    g1[2] = (int)((rowlen_elts >> 16) | (1u << 16));   // tensor_dim0[31:16] | tensor_dim1=1
    g1[3] = (int)((rowlen_elts & 0xFFFFu) << 16);      // tensor_dim1 hi=0 | tile_dim0
    g1[4] = 0;                                         // tile_dim1=0, tile_dim2=0 (1D tile)
    g1[5] = (int)grow_stride_elts;                     // tensor_dim0_stride[31:0]
    g1[6] = 0;
    g1[7] = 0;
    v4i g2;
    g2[0] = 0;                                         // tensor_dim2 (unused)
    g2[1] = (int)lds_stride_elts;                      // lds_addr_increment per iter
    g2[2] = (int)grow_stride_elts;                     // global_addr_increment[31:0]
    g2[3] = (int)((nrows - 1u) << 16);                 // iterate_count (0 => 1x)
    v4i g3 = (v4i){0, 0, 0, 0};
    v8i g4 = (v8i){0, 0, 0, 0, 0, 0, 0, 0};
    __builtin_amdgcn_tensor_load_to_lds(g0, g1, g2, g3, g4, 0);
}
#endif

// ---------------------------------------------------------------------------
// Small utility kernels
// ---------------------------------------------------------------------------
__global__ void zero_ctr_kernel(unsigned* ctr) {
    if (threadIdx.x == 0) *ctr = 0u;
}

__global__ void cvt_f32_bf16_kernel(const float* __restrict__ in,
                                    __bf16* __restrict__ out, int n) {
    for (int i = blockIdx.x * blockDim.x + threadIdx.x; i < n;
         i += gridDim.x * blockDim.x)
        out[i] = (__bf16)in[i];
}

// ---------------------------------------------------------------------------
// Kernel 1: A_all[t,b,h] = sum_i x[t,b,i] * WA[h,i]
// WG = 256 threads = 8 waves; block tile 128(M) x 64(N); wave tile 32x32
// (2x2 WMMA tiles); K streamed in chunks of 32 (bf16).
// ---------------------------------------------------------------------------
__global__ __launch_bounds__(256)
void gemm_A_kernel(const float* __restrict__ X,
                   const __bf16* __restrict__ WAbf,
                   float* __restrict__ Aall) {
    const int lane  = threadIdx.x & 31;
    const int wave  = threadIdx.x >> 5;
    const int wrow  = wave >> 1;          // 0..3
    const int wcol  = wave & 1;           // 0..1
    const int nlane = lane & 15;
    const int mhalf = (lane < 16) ? 0 : 8;
    const int am    = lane & 15;
    const int kb1   = (lane < 16) ? 0 : 8;
    const int kb2   = kb1 + 16;

    const int r0 = blockIdx.x * 128 + wrow * 32;   // M base (t*B+b flattened)
    const int c0 = blockIdx.y * 64  + wcol * 32;   // N base (h)

    const float*  arow0 = X    + (size_t)(r0 + am)        * K_DIM;
    const float*  arow1 = X    + (size_t)(r0 + 16 + am)   * K_DIM;
    const __bf16* brow0 = WAbf + (size_t)(c0 + nlane)     * K_DIM;
    const __bf16* brow1 = WAbf + (size_t)(c0 + 16 + nlane)* K_DIM;

    v8f acc00 = {}, acc01 = {}, acc10 = {}, acc11 = {};

    for (int k0 = 0; k0 < K_DIM; k0 += 32) {
        __builtin_prefetch(brow0 + k0 + 64, 0, 1);
        __builtin_prefetch(brow1 + k0 + 64, 0, 1);
        v16bf a0 = ld_frag_f32(arow0, k0 + kb1, k0 + kb2);
        v16bf a1 = ld_frag_f32(arow1, k0 + kb1, k0 + kb2);
        v16bf b0 = ld_frag_bf16(brow0, k0 + kb1, k0 + kb2);
        v16bf b1 = ld_frag_bf16(brow1, k0 + kb1, k0 + kb2);
        acc00 = wmma_bf16(a0, b0, acc00);
        acc01 = wmma_bf16(a0, b1, acc01);
        acc10 = wmma_bf16(a1, b0, acc10);
        acc11 = wmma_bf16(a1, b1, acc11);
    }

    // C/D layout: VGPR r <-> row (r + mhalf), lane%16 <-> col
    v8f* accs[4] = { &acc00, &acc01, &acc10, &acc11 };
    #pragma unroll
    for (int i = 0; i < 2; ++i)
        #pragma unroll
        for (int j = 0; j < 2; ++j) {
            v8f& a = *accs[i * 2 + j];
            float* out = Aall + (size_t)(r0 + i * 16 + mhalf) * HD_DIM
                              + (c0 + j * 16 + nlane);
            #pragma unroll
            for (int r = 0; r < 8; ++r)
                out[(size_t)r * HD_DIM] = a[r];
        }
}

// ---------------------------------------------------------------------------
// Device-wide barrier: monotonic counter, one arrival per WG per epoch.
// ---------------------------------------------------------------------------
__device__ __forceinline__ void grid_barrier(unsigned* ctr, unsigned target) {
    __syncthreads();
    if (threadIdx.x == 0) {
        __threadfence();
        __hip_atomic_fetch_add(ctr, 1u, __ATOMIC_RELEASE, __HIP_MEMORY_SCOPE_AGENT);
        while (__hip_atomic_load(ctr, __ATOMIC_ACQUIRE, __HIP_MEMORY_SCOPE_AGENT) < target)
            __builtin_amdgcn_s_sleep(2);
    }
    __syncthreads();
    __threadfence();
}

// ---------------------------------------------------------------------------
// Kernel 2: persistent scan. 32 WGs x 256 threads. Each WG owns 32 paired
// (F,R) columns. The 128 KB weight slice is staged ONCE into LDS via the
// Tensor Data Mover (padded row stride -> conflict-free ds_load_b128), so the
// 512-step inner loop reads B-fragments from LDS and only the 128 KB V buffer
// (L2-resident) from global. S lives in registers; one grid barrier per step.
// ---------------------------------------------------------------------------
__global__ __launch_bounds__(256)
void scan_kernel(const float* __restrict__ X,   const float* __restrict__ Q0,
                 const float* __restrict__ S0,
                 const float* __restrict__ uV_raw, const float* __restrict__ uV2,
                 const float* __restrict__ PFB, const float* __restrict__ HWb,
                 const __bf16* __restrict__ Wbf, const float* __restrict__ Aall,
                 __bf16* __restrict__ Vbuf, unsigned* __restrict__ ctr,
                 float* __restrict__ Y, float* __restrict__ Qf,
                 float* __restrict__ Sf) {
    extern __shared__ __bf16 ldsW[];           // [F: 32 rows][R: 32 rows], stride 1032
    const unsigned ldsRegionR = 32u * LDSW_STRIDE;

    const int tid   = threadIdx.x;
    const int lane  = tid & 31;
    const int wave  = tid >> 5;
    const int h0    = blockIdx.x * HS;
    const int wrow  = wave >> 1;              // 0..3 -> b block
    const int wcol  = wave & 1;               // 0..1 -> col block
    const int b0    = wrow * 16;
    const int c0    = h0 + wcol * 16;
    const int nlane = lane & 15;
    const int mhalf = (lane < 16) ? 0 : 8;
    const int am    = lane & 15;
    const int kb1   = (lane < 16) ? 0 : 8;
    const int kb2   = kb1 + 16;
    const int hcol  = c0 + nlane;             // the one column this lane owns

    // ---- stage weight slice into LDS (once) --------------------------------
#if defined(HAVE_TDM)
    if (wave == 0) {
        unsigned base = (unsigned)(uintptr_t)(void*)&ldsW[0];
        tdm_load_rows(base, Wbf + (size_t)h0 * K_DIM,
                      32, K_DIM, K_DIM, LDSW_STRIDE);
        tdm_load_rows(base + ldsRegionR * 2u, Wbf + (size_t)(HD_DIM + h0) * K_DIM,
                      32, K_DIM, K_DIM, LDSW_STRIDE);
        __builtin_amdgcn_s_wait_tensorcnt(0);
    }
#else
    for (int e = tid * 8; e < 32 * K_DIM; e += 256 * 8) {
        int j = e / K_DIM, k = e % K_DIM;
        *reinterpret_cast<uint4*>(&ldsW[(size_t)j * LDSW_STRIDE + k]) =
            *reinterpret_cast<const uint4*>(&Wbf[(size_t)(h0 + j) * K_DIM + k]);
        *reinterpret_cast<uint4*>(&ldsW[ldsRegionR + (size_t)j * LDSW_STRIDE + k]) =
            *reinterpret_cast<const uint4*>(&Wbf[(size_t)(HD_DIM + h0 + j) * K_DIM + k]);
    }
#endif

    // ---- init V for t=0 over this WG's i-slice -----------------------------
    for (int e = tid; e < B_DIM * HS; e += 256) {
        int b = e / HS;
        int i = h0 + (e % HS);
        float g = sigmoidf_(uV2[i]) * tanhf(uV_raw[i]);
        Vbuf[(size_t)b * HD_DIM + i] =
            (__bf16)(X[(size_t)b * HD_DIM + i] + g * Q0[(size_t)b * HD_DIM + i]);
    }

    const float gate = sigmoidf_(uV2[hcol]) * tanhf(uV_raw[hcol]);
    const float pfb  = PFB[hcol];
    const float hwb  = HWb[hcol];

    float s[8];
    #pragma unroll
    for (int r = 0; r < 8; ++r)
        s[r] = S0[(size_t)(b0 + mhalf + r) * HD_DIM + hcol];

    __syncthreads();                          // LDS weights visible to all waves
    unsigned epoch = 1;
    grid_barrier(ctr, epoch * NWG);           // V(t=0) published device-wide

    const __bf16* arow  = Vbuf + (size_t)(b0 + am) * HD_DIM;
    const int     jcol  = wcol * 16 + nlane;  // LDS weight row for this lane
    const __bf16* lrowF = &ldsW[(size_t)jcol * LDSW_STRIDE];
    const __bf16* lrowR = &ldsW[ldsRegionR + (size_t)jcol * LDSW_STRIDE];

    for (int t = 0; t < T_STEPS; ++t) {
        v8f accF = {}, accR = {};
        #pragma unroll 4
        for (int k0 = 0; k0 < K_DIM; k0 += 32) {
            v16bf a  = ld_frag_bf16(arow,  k0 + kb1, k0 + kb2);
            v16bf bF = ld_frag_bf16(lrowF, k0 + kb1, k0 + kb2);
            v16bf bR = ld_frag_bf16(lrowR, k0 + kb1, k0 + kb2);
            accF = wmma_bf16(a, bF, accF);
            accR = wmma_bf16(a, bR, accR);
        }

        const float* xrow  = X    + (size_t)t * B_DIM * HD_DIM;
        const float* arowA = Aall + (size_t)t * B_DIM * HD_DIM;
        float*       yrow  = Y    + (size_t)t * B_DIM * HD_DIM;

        float qn[8];
        #pragma unroll
        for (int r = 0; r < 8; ++r) {
            int b = b0 + mhalf + r;
            size_t off = (size_t)b * HD_DIM + hcol;
            float F  = sigmoidf_(accF[r] + pfb);
            float R  = sigmoidf_(accR[r] + hwb);
            float sn = F * s[r] + (1.0f - F) * arowA[off];
            float q  = R * sn + (1.0f - R) * xrow[off];
            s[r] = sn;
            qn[r] = q;
            yrow[off] = q;
        }

        if (t + 1 < T_STEPS) {
            // Build next V directly from registers (Q slice is column-local).
            const float* xn = xrow + (size_t)B_DIM * HD_DIM;
            #pragma unroll
            for (int r = 0; r < 8; ++r) {
                int b = b0 + mhalf + r;
                size_t off = (size_t)b * HD_DIM + hcol;
                Vbuf[off] = (__bf16)(xn[off] + gate * qn[r]);
            }
        } else {
            #pragma unroll
            for (int r = 0; r < 8; ++r) {
                int b = b0 + mhalf + r;
                size_t off = (size_t)b * HD_DIM + hcol;
                Qf[off] = qn[r];
                Sf[off] = s[r];
            }
        }

        ++epoch;
        grid_barrier(ctr, epoch * NWG);       // V(t+1) published
    }
}

// ---------------------------------------------------------------------------
// Host entry
// ---------------------------------------------------------------------------
extern "C" void kernel_launch(void* const* d_in, const int* in_sizes, int n_in,
                              void* d_out, int out_size, void* d_ws, size_t ws_size,
                              hipStream_t stream) {
    (void)in_sizes; (void)n_in; (void)out_size; (void)ws_size;

    const float* x      = (const float*)d_in[0];
    const float* Q0     = (const float*)d_in[1];
    const float* S0     = (const float*)d_in[2];
    const float* Wih    = (const float*)d_in[3];
    const float* WA     = (const float*)d_in[4];
    const float* uV_raw = (const float*)d_in[5];
    const float* uV2    = (const float*)d_in[6];
    const float* PFB    = (const float*)d_in[7];
    const float* HWb    = (const float*)d_in[8];

    char* ws = (char*)d_ws;
    __bf16*   Wbf  = (__bf16*)(ws + 0);                    // 4 MiB (2HD x K bf16)
    __bf16*   WAbf = (__bf16*)(ws + ((size_t)4  << 20));   // 2 MiB
    __bf16*   Vbuf = (__bf16*)(ws + ((size_t)6  << 20));   // 128 KiB
    unsigned* ctr  = (unsigned*)(ws + ((size_t)7 << 20));  // barrier counter
    float*    Aall = (float*)  (ws + ((size_t)8  << 20));  // 128 MiB

    float* Y  = (float*)d_out;
    float* Qf = Y  + (size_t)T_STEPS * B_DIM * HD_DIM;
    float* Sf = Qf + (size_t)B_DIM * HD_DIM;

    zero_ctr_kernel<<<1, 32, 0, stream>>>(ctr);
    cvt_f32_bf16_kernel<<<2048, 256, 0, stream>>>(Wih, Wbf, 2 * HD_DIM * K_DIM);
    cvt_f32_bf16_kernel<<<1024, 256, 0, stream>>>(WA, WAbf, HD_DIM * K_DIM);

    dim3 grid_gemm((T_STEPS * B_DIM) / 128, HD_DIM / 64);
    gemm_A_kernel<<<grid_gemm, 256, 0, stream>>>(x, WAbf, Aall);

    scan_kernel<<<NWG, 256, SCAN_LDS_BYTES, stream>>>(x, Q0, S0, uV_raw, uV2,
                                                      PFB, HWb, Wbf, Aall, Vbuf,
                                                      ctr, Y, Qf, Sf);
}
